// Attention_Module_35596688949498
// MI455X (gfx1250) — matrix operand order repaired and checked by reading, not compiled, not run on hardware
//
#include <hip/hip_runtime.h>

// ---------------------------------------------------------------------------
// Types for WMMA fragments (wave32, V_WMMA_F32_16X16X32_F16)
// ---------------------------------------------------------------------------
typedef _Float16 v4h  __attribute__((ext_vector_type(4)));
typedef _Float16 v8h  __attribute__((ext_vector_type(8)));
typedef _Float16 v16h __attribute__((ext_vector_type(16)));
typedef float    v8f  __attribute__((ext_vector_type(8)));

#define WMMA_F16(a, b, c) \
  __builtin_amdgcn_wmma_f32_16x16x32_f16(false, (a), false, (b), (short)0, (c), false, false)

// A-fragment (16x32 f16): lane M = lane&15; halves j -> K = hi*16 + j
// (hi = lane>>4). Row must be K-contiguous.
__device__ __forceinline__ v16h load_A16(const _Float16* row, int hi) {
  v8h lo = *reinterpret_cast<const v8h*>(row + hi * 16);
  v8h hv = *reinterpret_cast<const v8h*>(row + hi * 16 + 8);
  v16h a;
#pragma unroll
  for (int i = 0; i < 8; ++i) { a[i] = lo[i]; a[i + 8] = hv[i]; }
  return a;
}

// B-fragment (32x16 f16): lane N = lane&15; halves j -> K = hi*16 + j.
// "row" points at the 32 K-contiguous halves of column N.
__device__ __forceinline__ v16h load_B16(const _Float16* row, int hi) {
  return *reinterpret_cast<const v16h*>(row + hi * 16);
}

__device__ __forceinline__ unsigned pack2(float a, float b) {
  union { _Float16 h[2]; unsigned u; } z;
  z.h[0] = (_Float16)a; z.h[1] = (_Float16)b;
  return z.u;
}

// ---------------------------------------------------------------------------
// CDNA5 async memory->LDS copy (ASYNCcnt-tracked), per-lane 16B.
// VDST = LDS byte offset (low 32 bits of the LDS aperture address),
// VADDR = 64-bit global address, saddr = off (null).
// ---------------------------------------------------------------------------
__device__ __forceinline__ void async_copy16(const _Float16* lds_ptr, const _Float16* g) {
  unsigned lds_off = (unsigned)(unsigned long long)lds_ptr;
  unsigned long long ga = (unsigned long long)g;
  asm volatile("global_load_async_to_lds_b128 %0, %1, off"
               :: "v"(lds_off), "v"(ga)
               : "memory");
}

__device__ __forceinline__ void wait_async0() {
  asm volatile("s_wait_asynccnt 0x0" ::: "memory");
}

// ---------------------------------------------------------------------------
// K0: fold BN into f16 weights + f32 bias.  W' = w*scale, b' = (b-m)*scale+t
// ---------------------------------------------------------------------------
__global__ void fold_bn_kernel(const float* __restrict__ w, const float* __restrict__ b,
                               const float* __restrict__ s, const float* __restrict__ t,
                               const float* __restrict__ m, const float* __restrict__ v,
                               _Float16* __restrict__ w16, float* __restrict__ bias,
                               int OC, int IC) {
  int i = blockIdx.x * 256 + threadIdx.x;
  if (i < OC * IC) {
    int oc = i / IC;
    float sc = s[oc] * rsqrtf(v[oc] + 1e-5f);
    w16[i] = (_Float16)(w[i] * sc);
  }
  if (i < OC) {
    float sc = s[i] * rsqrtf(v[i] + 1e-5f);
    bias[i] = (b[i] - m[i]) * sc + t[i];
  }
}

// ---------------------------------------------------------------------------
// K1: 1x1 conv + folded-BN (+ReLU) with WMMA.  Block = 8 waves = 32oc x 128px
// (= two image rows, so 2x2 maxpool is block-local).  K stepped by 32.
// EPI: 0 = f   (ReLU, pool,   store (n,k)  f16 [B][1024][32])
//      1 = g   (ReLU, nopool, store (m,k)  f16 [B][4096][32])
//      2 = hh  (ReLU, pool,   store (c,n)  f16 [B][128][1024])
//      3 = out (no ReLU, f16 (px,c) input, fp32 gamma*y + x residual)
// ---------------------------------------------------------------------------
template <int CIN, int EPI>
__launch_bounds__(256)
__global__ void conv1x1_bn_kernel(const float* __restrict__ xin,        // [B][CIN][4096] (EPI 0..2)
                                  const _Float16* __restrict__ xin_t,   // [B][4096][CIN] (EPI 3)
                                  const _Float16* __restrict__ w16,     // [OC][CIN]
                                  const float* __restrict__ bias,       // [OC]
                                  _Float16* __restrict__ yout,
                                  const float* __restrict__ xres,
                                  const float* __restrict__ gamma_p,
                                  float* __restrict__ out32) {
  constexpr int HW = 4096;
  __shared__ _Float16 __align__(32) Bs[128][32];  // x tile, (px, k) f16
  __shared__ float Ys[128][33];                   // epilogue tile, (px, oc) f32

  const int lane = threadIdx.x & 31;
  const int wave = threadIdx.x >> 5;
  const int hi = lane >> 4;
  const int l15 = lane & 15;
  const int mi = wave >> 2;   // 0..1 : oc half-tile
  const int ni = wave & 3;    // 0..3 : 32-px slab
  const int b = blockIdx.z;
  const int pxbase = blockIdx.x * 128;
  const int octile = blockIdx.y * 32;

  v8f acc0 = {}; v8f acc1 = {};
  const _Float16* wrow = w16 + (size_t)(octile + mi * 16 + l15) * CIN;

  for (int kk = 0; kk < CIN; kk += 32) {
    v16h b0, b1;
    if (EPI != 3) {
      __syncthreads();
      const int ch = threadIdx.x >> 3;          // 0..31
      const int px0 = (threadIdx.x & 7) * 16;   // 16 px each, coalesced f32 reads
      const float* src = xin + ((size_t)b * CIN + (kk + ch)) * HW + pxbase + px0;
      if (kk + 32 < CIN) __builtin_prefetch(src + 32 * HW, 0, 1);
#pragma unroll
      for (int i = 0; i < 16; ++i) Bs[px0 + i][ch] = (_Float16)src[i];
      __syncthreads();
      b0 = *reinterpret_cast<const v16h*>(&Bs[ni * 32 + l15][hi * 16]);
      b1 = *reinterpret_cast<const v16h*>(&Bs[ni * 32 + 16 + l15][hi * 16]);
    } else {
      const _Float16* base = xin_t + ((size_t)b * HW + pxbase) * CIN + kk;
      b0 = load_B16(base + (size_t)(ni * 32 + l15) * CIN, hi);
      b1 = load_B16(base + (size_t)(ni * 32 + 16 + l15) * CIN, hi);
    }
    v16h a = load_A16(wrow + kk, hi);
    acc0 = WMMA_F16(a, b0, acc0);
    acc1 = WMMA_F16(a, b1, acc1);
  }

  // D layout: lane col N = l15 (px), rows M = hi*8 + v (oc). Dump to LDS.
  const int ocl = mi * 16 + hi * 8;
  const int pxl0 = ni * 32 + l15;
  const int pxl1 = ni * 32 + 16 + l15;
#pragma unroll
  for (int v = 0; v < 8; ++v) {
    float bb = bias[octile + ocl + v];
    float y0 = acc0[v] + bb;
    float y1 = acc1[v] + bb;
    if (EPI != 3) { y0 = fmaxf(y0, 0.f); y1 = fmaxf(y1, 0.f); }
    Ys[pxl0][ocl + v] = y0;
    Ys[pxl1][ocl + v] = y1;
  }
  __syncthreads();

  if (EPI == 0) {                         // pooled, (n, k) f16
    const int nl = threadIdx.x >> 3;      // pooled col 0..31
    const int og = (threadIdx.x & 7) * 4; // 4 oc each
    const int c0 = 2 * nl;
    v4h pk;
#pragma unroll
    for (int j = 0; j < 4; ++j) {
      float m0 = fmaxf(Ys[c0][og + j], Ys[c0 + 1][og + j]);
      float m1 = fmaxf(Ys[64 + c0][og + j], Ys[65 + c0][og + j]);
      pk[j] = (_Float16)fmaxf(m0, m1);
    }
    const size_t n = (size_t)(pxbase >> 7) * 32 + nl;
    *reinterpret_cast<v4h*>(yout + ((size_t)b * 1024 + n) * 32 + og) = pk;
  } else if (EPI == 1) {                  // (m, k) f16
    const int px = threadIdx.x >> 1;
    const int og = (threadIdx.x & 1) * 16;
    v16h pk;
#pragma unroll
    for (int i = 0; i < 16; ++i) pk[i] = (_Float16)Ys[px][og + i];
    *reinterpret_cast<v16h*>(yout + ((size_t)b * 4096 + pxbase + px) * 32 + og) = pk;
  } else if (EPI == 2) {                  // pooled, (c, n) f16
    const int ocr = threadIdx.x >> 3;     // 0..31
    const int ng = (threadIdx.x & 7) * 4; // 4 pooled cols each
    v4h pk;
#pragma unroll
    for (int j = 0; j < 4; ++j) {
      const int c0 = 2 * (ng + j);
      float m0 = fmaxf(Ys[c0][ocr], Ys[c0 + 1][ocr]);
      float m1 = fmaxf(Ys[64 + c0][ocr], Ys[65 + c0][ocr]);
      pk[j] = (_Float16)fmaxf(m0, m1);
    }
    const size_t n = (size_t)(pxbase >> 7) * 32 + ng;
    *reinterpret_cast<v4h*>(yout + ((size_t)b * 128 + octile + ocr) * 1024 + n) = pk;
  } else {                                // final: gamma*y + x, fp32
    const float g = gamma_p[0];
    const int ocr = threadIdx.x >> 3;
    const int pg = (threadIdx.x & 7) * 16;
    const size_t base = ((size_t)b * 256 + octile + ocr) * (size_t)HW + pxbase;
#pragma unroll
    for (int j = 0; j < 16; ++j)
      out32[base + pg + j] = g * Ys[pg + j][ocr] + xres[base + pg + j];
  }
}

// ---------------------------------------------------------------------------
// K2: flash-attention over n=1024, d_k=32, d_v=128, softmax over n.
// One wave owns 16 m-columns; O accumulator = 8 c-tiles (64 VGPRs/lane).
// Per 32-key chunk the block stages hh (8KB) + f (2KB) chunks into a double-
// buffered LDS tile with GLOBAL_LOAD_ASYNC_TO_LDS_B128 (ASYNCcnt), then each
// wave runs 2 score WMMAs + 8 accumulate WMMAs from LDS fragments.
// ---------------------------------------------------------------------------
__launch_bounds__(128)
__global__ void attn_kernel(const _Float16* __restrict__ f_t,  // [B][1024][32]
                            const _Float16* __restrict__ g_t,  // [B][4096][32]
                            const _Float16* __restrict__ hh,   // [B][128][1024]
                            _Float16* __restrict__ o_t) {      // [B][4096][128]
  __shared__ _Float16 __align__(16) hhs[2][128][32];  // (c, n-chunk) 2 x 8KB
  __shared__ _Float16 __align__(16) fs[2][32][32];    // (n-chunk, k) 2 x 2KB

  const int lane = threadIdx.x & 31;
  const int wave = threadIdx.x >> 5;
  const int hi = lane >> 4;
  const int l15 = lane & 15;
  const int b = blockIdx.y;
  const int m0 = (blockIdx.x * 4 + wave) * 16;
  const int t = threadIdx.x;

  // Query fragment (B operand of score GEMM), fixed for the whole sweep.
  v16h gB = load_B16(g_t + ((size_t)b * 4096 + m0 + l15) * 32, hi);

  const _Float16* fb = f_t + (size_t)b * 1024 * 32;
  const _Float16* hb = hh + (size_t)b * 128 * 1024;

  // issue one chunk's async copies (block-cooperative, 16B per lane per issue)
  auto issue = [&](int buf, int nc) {
    // hh chunk: 128 rows x 64B = 512 segments of 16B, 4 per thread
#pragma unroll
    for (int i = 0; i < 4; ++i) {
      int idx = t + i * 128;
      int c = idx >> 2, seg = idx & 3;
      async_copy16(&hhs[buf][c][seg * 8], hb + (size_t)c * 1024 + nc + seg * 8);
    }
    // f chunk: 32 rows x 64B = 128 segments of 16B, 1 per thread
    int n = t >> 2, seg = t & 3;
    async_copy16(&fs[buf][n][seg * 8], fb + (size_t)(nc + n) * 32 + seg * 8);
  };

  issue(0, 0);

  v8f O[8];
#pragma unroll
  for (int i = 0; i < 8; ++i) O[i] = (v8f){};
  float mrun = -1.0e30f, srun = 0.f;

  for (int nc = 0; nc < 1024; nc += 32) {
    const int cur = (nc >> 5) & 1;
    wait_async0();        // this wave's copies for `cur` are in LDS
    __syncthreads();      // ... and everyone else's too
    if (nc + 32 < 1024) issue(cur ^ 1, nc + 32);

    v16h a0 = load_A16(&fs[cur][l15][0], hi);
    v16h a1 = load_A16(&fs[cur][16 + l15][0], hi);
    v8f z = {};
    v8f s0 = WMMA_F16(a0, gB, z);
    v8f s1 = WMMA_F16(a1, gB, z);

    // column-wise (over n) running max: lane + lane^16 hold complementary rows
    float cm = s0[0];
#pragma unroll
    for (int v = 1; v < 8; ++v) cm = fmaxf(cm, s0[v]);
#pragma unroll
    for (int v = 0; v < 8; ++v) cm = fmaxf(cm, s1[v]);
    cm = fmaxf(cm, __shfl_xor(cm, 16, 32));
    const float mnew = fmaxf(mrun, cm);
    const float alpha = __expf(mrun - mnew);
    mrun = mnew;
    srun *= alpha;

    // p = exp(s - max), packed to f16 pairs; accumulate per-lane partial sum
    unsigned pk0[4], pk1[4];
    float psum = 0.f;
#pragma unroll
    for (int v = 0; v < 4; ++v) {
      float e0 = __expf(s0[2 * v] - mnew), e1 = __expf(s0[2 * v + 1] - mnew);
      float e2 = __expf(s1[2 * v] - mnew), e3 = __expf(s1[2 * v + 1] - mnew);
      psum += (e0 + e1) + (e2 + e3);
      pk0[v] = pack2(e0, e1);
      pk1[v] = pack2(e2, e3);
    }
    srun += psum;

    // Reassemble p into the B-fragment layout (lane m = l15, halves = n):
    // hi==0 needs [own p0 | partner p0], hi==1 needs [partner p1 | own p1].
    union { v16h v; unsigned u[8]; } pB;
#pragma unroll
    for (int v = 0; v < 4; ++v) {
      unsigned send = hi ? pk0[v] : pk1[v];
      unsigned recv = __shfl_xor(send, 16, 32);
      pB.u[v]     = hi ? recv : pk0[v];
      pB.u[v + 4] = hi ? pk1[v] : recv;
    }

    // rescale O by alpha (uniform per column pair of lanes) and accumulate
#pragma unroll
    for (int ct = 0; ct < 8; ++ct) {
#pragma unroll
      for (int v = 0; v < 8; ++v) O[ct][v] *= alpha;
      v16h ah = load_A16(&hhs[cur][ct * 16 + l15][0], hi);
      O[ct] = WMMA_F16(ah, pB.v, O[ct]);
    }
  }

  const float stot = srun + __shfl_xor(srun, 16, 32);
  const float inv = 1.0f / stot;

  // store o^T: row m = m0 + l15, c = ct*16 + hi*8 + v  (16B packed stores)
  _Float16* orow = o_t + ((size_t)b * 4096 + m0 + l15) * 128;
#pragma unroll
  for (int ct = 0; ct < 8; ++ct) {
    v8h pk;
#pragma unroll
    for (int v = 0; v < 8; ++v) pk[v] = (_Float16)(O[ct][v] * inv);
    *reinterpret_cast<v8h*>(orow + ct * 16 + hi * 8) = pk;
  }
}

// ---------------------------------------------------------------------------
// Launcher
// ---------------------------------------------------------------------------
extern "C" void kernel_launch(void* const* d_in, const int* in_sizes, int n_in,
                              void* d_out, int out_size, void* d_ws, size_t ws_size,
                              hipStream_t stream) {
  (void)in_sizes; (void)n_in; (void)out_size; (void)ws_size;
  const float* x = (const float*)d_in[0];
  auto IN = [&](int i) { return (const float*)d_in[i]; };

  char* ws = (char*)d_ws;
  size_t off = 0;
  auto alloc = [&](size_t bytes) -> void* {
    void* p = ws + off;
    off += (bytes + 255) & ~(size_t)255;
    return p;
  };

  _Float16* W1 = (_Float16*)alloc((size_t)32 * 256 * 2);
  _Float16* W2 = (_Float16*)alloc((size_t)32 * 256 * 2);
  _Float16* W3 = (_Float16*)alloc((size_t)128 * 256 * 2);
  _Float16* W4 = (_Float16*)alloc((size_t)256 * 128 * 2);
  float* B1 = (float*)alloc(32 * 4);
  float* B2 = (float*)alloc(32 * 4);
  float* B3 = (float*)alloc(128 * 4);
  float* B4 = (float*)alloc(256 * 4);
  _Float16* Ft = (_Float16*)alloc((size_t)8 * 1024 * 32 * 2);
  _Float16* Gt = (_Float16*)alloc((size_t)8 * 4096 * 32 * 2);
  _Float16* Hh = (_Float16*)alloc((size_t)8 * 128 * 1024 * 2);
  _Float16* Ot = (_Float16*)alloc((size_t)8 * 4096 * 128 * 2);

  // fold BN into f16 weights
  fold_bn_kernel<<<dim3((32 * 256 + 255) / 256), dim3(256), 0, stream>>>(
      IN(1), IN(2), IN(3), IN(4), IN(5), IN(6), W1, B1, 32, 256);
  fold_bn_kernel<<<dim3((32 * 256 + 255) / 256), dim3(256), 0, stream>>>(
      IN(7), IN(8), IN(9), IN(10), IN(11), IN(12), W2, B2, 32, 256);
  fold_bn_kernel<<<dim3((128 * 256 + 255) / 256), dim3(256), 0, stream>>>(
      IN(13), IN(14), IN(15), IN(16), IN(17), IN(18), W3, B3, 128, 256);
  fold_bn_kernel<<<dim3((256 * 128 + 255) / 256), dim3(256), 0, stream>>>(
      IN(19), IN(20), IN(21), IN(22), IN(23), IN(24), W4, B4, 256, 128);

  // f, g, hh projections (grid: px-tiles of 128, oc-tiles of 32, batch)
  conv1x1_bn_kernel<256, 1><<<dim3(32, 1, 8), 256, 0, stream>>>(
      x, nullptr, W2, B2, Gt, nullptr, nullptr, nullptr);
  conv1x1_bn_kernel<256, 0><<<dim3(32, 1, 8), 256, 0, stream>>>(
      x, nullptr, W1, B1, Ft, nullptr, nullptr, nullptr);
  conv1x1_bn_kernel<256, 2><<<dim3(32, 4, 8), 256, 0, stream>>>(
      x, nullptr, W3, B3, Hh, nullptr, nullptr, nullptr);

  // attention (flash style over n=1024; never materializes s)
  attn_kernel<<<dim3(64, 8), 128, 0, stream>>>(Ft, Gt, Hh, Ot);

  // final conv + BN + gamma*y + x
  conv1x1_bn_kernel<128, 3><<<dim3(32, 8, 8), 256, 0, stream>>>(
      nullptr, Ot, W4, B4, nullptr, x, IN(25), (float*)d_out);
}
